// DAGNN_52261162057822
// MI455X (gfx1250) — compile-verified, matrix operand order checked by reading, not run on hardware
//
#include <hip/hip_runtime.h>
#include <math.h>

typedef __attribute__((ext_vector_type(16))) __bf16 v16bf;
typedef __attribute__((ext_vector_type(8)))  __bf16 v8bf;
typedef __attribute__((ext_vector_type(4)))  __bf16 v4bf;
typedef __attribute__((ext_vector_type(8)))  float  v8f;

// ---------------------------------------------------------------------------
// helpers
// ---------------------------------------------------------------------------
__device__ __forceinline__ void splitf(float f, __bf16& hi, __bf16& lo) {
    hi = (__bf16)f;
    lo = (__bf16)(f - (float)hi);
}

// Fragment loader for 16-bit WMMA operands (ISA 7.12.2 layout).
// lane l -> row/col (l&15); K-halves by (l>>4): elems 0-7 = K+{0..7 | 8..15},
// elems 8-15 = K+{16..23 | 24..31}. Works for A (base in LDS, idx0=0) and
// B (pre-transposed [Ncol][K] weights in global, idx0=n0).
__device__ __forceinline__ v16bf load_frag(const __bf16* base, int ldk, int idx0, int k0) {
    int lane = threadIdx.x & 31;
    int i  = lane & 15;
    int kb = (lane >> 4) * 8;
    const __bf16* p = base + (size_t)(idx0 + i) * ldk + k0 + kb;
    v8bf a = *(const v8bf*)p;
    v8bf b = *(const v8bf*)(p + 16);
    v16bf r;
#pragma unroll
    for (int j = 0; j < 8; ++j) { r[j] = a[j]; r[8 + j] = b[j]; }
    return r;
}

__device__ __forceinline__ v8f wmma_bf16(v16bf a, v16bf b, v8f c) {
    return __builtin_amdgcn_wmma_f32_16x16x32_bf16(false, a, false, b,
                                                   (short)0, c, false, false);
}

// ---------------------------------------------------------------------------
// weight prep: split f32 [K][Ncol] into bf16 hi/lo, transposed to [Ncol][K]
// ---------------------------------------------------------------------------
__global__ void split_w_kernel(const float* __restrict__ w, int K, int Ncol,
                               __bf16* __restrict__ hi, __bf16* __restrict__ lo) {
    int idx = blockIdx.x * 256 + threadIdx.x;
    if (idx >= K * Ncol) return;
    int k = idx / Ncol, n = idx % Ncol;
    float f = w[idx];
    __bf16 h, l; splitf(f, h, l);
    hi[(size_t)n * K + k] = h;
    lo[(size_t)n * K + k] = l;
}

// ---------------------------------------------------------------------------
// fused 2-layer MLP: h0 = relu(x@W1+b1)@W2+b2   (bf16x3 split-precision WMMA)
// one block = 16 nodes, 8 waves. x is converted to bf16 hi/lo ONCE during
// LDS staging; all WMMA operands are then raw ds/global b128 loads.
// ---------------------------------------------------------------------------
__global__ __launch_bounds__(256) void mlp_kernel(
    const float* __restrict__ x,
    const __bf16* __restrict__ w1hi, const __bf16* __restrict__ w1lo,
    const float* __restrict__ b1,
    const __bf16* __restrict__ w2hi, const __bf16* __restrict__ w2lo,
    const float* __restrict__ b2,
    float* __restrict__ h0, int N) {
    __shared__ __align__(16) __bf16 xhi[16 * 512];   // 16 KB
    __shared__ __align__(16) __bf16 xlo[16 * 512];   // 16 KB
    __shared__ __align__(16) __bf16 h1hi[16 * 256];  //  8 KB
    __shared__ __align__(16) __bf16 h1lo[16 * 256];  //  8 KB

    int nb = blockIdx.x * 16;

    // stage + split x block: float4 global loads, packed v4bf LDS stores
    {
        const float4* x4 = (const float4*)x;
        for (int i = threadIdx.x; i < 16 * 128; i += 256) {
            int r = i >> 7;
            float4 f = make_float4(0.f, 0.f, 0.f, 0.f);
            if (nb + r < N) f = x4[(size_t)nb * 128 + i];
            v4bf h4, l4;
            __bf16 h, l;
            splitf(f.x, h, l); h4[0] = h; l4[0] = l;
            splitf(f.y, h, l); h4[1] = h; l4[1] = l;
            splitf(f.z, h, l); h4[2] = h; l4[2] = l;
            splitf(f.w, h, l); h4[3] = h; l4[3] = l;
            *(v4bf*)(xhi + i * 4) = h4;
            *(v4bf*)(xlo + i * 4) = l4;
        }
    }
    __syncthreads();

    int wave = threadIdx.x >> 5;
    int lane = threadIdx.x & 31;
    int cn   = lane & 15;
    int mb   = (lane >> 4) * 8;

    // ---- GEMM1: [16x512] x [512x256], wave w -> cols [32w, 32w+32) ----
    v8f acc0 = {}; v8f acc1 = {};
    int n0 = wave * 32;
    for (int k0 = 0; k0 < 512; k0 += 32) {
        v16bf ahi = load_frag(xhi, 512, 0, k0);
        v16bf alo = load_frag(xlo, 512, 0, k0);
        v16bf b0h = load_frag(w1hi, 512, n0,      k0);
        v16bf b0l = load_frag(w1lo, 512, n0,      k0);
        v16bf b1h = load_frag(w1hi, 512, n0 + 16, k0);
        v16bf b1l = load_frag(w1lo, 512, n0 + 16, k0);
        acc0 = wmma_bf16(ahi, b0h, acc0);
        acc0 = wmma_bf16(ahi, b0l, acc0);
        acc0 = wmma_bf16(alo, b0h, acc0);
        acc1 = wmma_bf16(ahi, b1h, acc1);
        acc1 = wmma_bf16(ahi, b1l, acc1);
        acc1 = wmma_bf16(alo, b1h, acc1);
    }

    // relu(acc + b1) -> LDS, already split to bf16 hi/lo
    {
        float bias0 = b1[n0 + cn];
        float bias1 = b1[n0 + 16 + cn];
#pragma unroll
        for (int r = 0; r < 8; ++r) {
            float v0 = acc0[r] + bias0; v0 = v0 > 0.0f ? v0 : 0.0f;
            float v1 = acc1[r] + bias1; v1 = v1 > 0.0f ? v1 : 0.0f;
            __bf16 h, l;
            int o0 = (r + mb) * 256 + n0 + cn;
            int o1 = o0 + 16;
            splitf(v0, h, l); h1hi[o0] = h; h1lo[o0] = l;
            splitf(v1, h, l); h1hi[o1] = h; h1lo[o1] = l;
        }
    }
    __syncthreads();

    // ---- GEMM2: [16x256] x [256x64], waves 0-3 -> cols [16w, 16w+16) ----
    if (wave < 4) {
        v8f acc = {};
        int c0 = wave * 16;
        for (int k0 = 0; k0 < 256; k0 += 32) {
            v16bf ahi = load_frag(h1hi, 256, 0, k0);
            v16bf alo = load_frag(h1lo, 256, 0, k0);
            v16bf bh  = load_frag(w2hi, 256, c0, k0);
            v16bf bl  = load_frag(w2lo, 256, c0, k0);
            acc = wmma_bf16(ahi, bh, acc);
            acc = wmma_bf16(ahi, bl, acc);
            acc = wmma_bf16(alo, bh, acc);
        }
        float bias = b2[c0 + cn];
#pragma unroll
        for (int r = 0; r < 8; ++r) {
            int node = nb + r + mb;
            if (node < N) h0[(size_t)node * 64 + c0 + cn] = acc[r] + bias;
        }
    }
}

// ---------------------------------------------------------------------------
// graph normalization
// ---------------------------------------------------------------------------
__global__ void fill_kernel(float* __restrict__ p, float v, long long n) {
    long long i = (long long)blockIdx.x * 256 + threadIdx.x;
    if (i < n) p[i] = v;
}

__global__ void deg_kernel(const int* __restrict__ col, float* __restrict__ deg, int E) {
    int i = blockIdx.x * 256 + threadIdx.x;
    if (i < E) atomicAdd(&deg[col[i]], 1.0f);
}

__global__ void dinv_kernel(float* __restrict__ deg_inout, float* __restrict__ selfnorm, int N) {
    int i = blockIdx.x * 256 + threadIdx.x;
    if (i >= N) return;
    float d  = deg_inout[i];
    float di = d > 0.0f ? rsqrtf(d) : 0.0f;
    deg_inout[i] = di;
    selfnorm[i]  = di * di;
}

__global__ void norm_kernel(const int* __restrict__ row, const int* __restrict__ col,
                            const float* __restrict__ dinv, float* __restrict__ nrm, int E) {
    int i = blockIdx.x * 256 + threadIdx.x;
    if (i < E) nrm[i] = dinv[row[i]] * dinv[col[i]];
}

// h_next = selfnorm ⊙ h_prev  (self-loop term, also zero-initializes)
__global__ void selfinit_kernel(const float* __restrict__ hprev,
                                const float* __restrict__ selfnorm,
                                float* __restrict__ hnext, long long NC) {
    long long i = (long long)blockIdx.x * 256 + threadIdx.x;
    if (i < NC) hnext[i] = selfnorm[i >> 6] * hprev[i];
}

// h_next[col] += norm * h_prev[row]  — one lane per (edge, channel).
// e is wave-uniform (wave32; 64 lanes per edge, block multiple of 32):
// force scalar so index/norm loads are SGPR loads, not 32-lane broadcasts.
__global__ void scatter_kernel(const int* __restrict__ row, const int* __restrict__ col,
                               const float* __restrict__ nrm,
                               const float* __restrict__ hprev,
                               float* __restrict__ hnext, int E) {
    long long t = (long long)blockIdx.x * 256 + threadIdx.x;
    int e = __builtin_amdgcn_readfirstlane((int)(t >> 6));
    int c = (int)(t & 63);
    if (e >= E) return;
    float v = nrm[e] * hprev[(size_t)row[e] * 64 + c];
    atomicAdd(&hnext[(size_t)col[e] * 64 + c], v);
}

// out += sigmoid(h·w_proj + b_proj) ⊙ h   — one wave per node
__global__ __launch_bounds__(256) void retain_acc_kernel(
    const float* __restrict__ h, const float* __restrict__ wproj,
    const float* __restrict__ bproj, float* __restrict__ out, int N) {
    int wave = threadIdx.x >> 5, lane = threadIdx.x & 31;
    int n = blockIdx.x * 8 + wave;
    if (n >= N) return;
    float v0 = h[(size_t)n * 64 + lane];
    float v1 = h[(size_t)n * 64 + 32 + lane];
    float part = v0 * wproj[lane] + v1 * wproj[32 + lane];
#pragma unroll
    for (int m = 16; m > 0; m >>= 1) part += __shfl_xor(part, m, 32);
    float r = 1.0f / (1.0f + expf(-(part + bproj[0])));
    out[(size_t)n * 64 + lane]      += r * v0;
    out[(size_t)n * 64 + 32 + lane] += r * v1;
}

// log_softmax over 64 channels — one wave per node
__global__ __launch_bounds__(256) void lsm_kernel(const float* __restrict__ o,
                                                  float* __restrict__ out, int N) {
    int wave = threadIdx.x >> 5, lane = threadIdx.x & 31;
    int n = blockIdx.x * 8 + wave;
    if (n >= N) return;
    float v0 = o[(size_t)n * 64 + lane];
    float v1 = o[(size_t)n * 64 + 32 + lane];
    float mx = fmaxf(v0, v1);
#pragma unroll
    for (int m = 16; m > 0; m >>= 1) mx = fmaxf(mx, __shfl_xor(mx, m, 32));
    float s = expf(v0 - mx) + expf(v1 - mx);
#pragma unroll
    for (int m = 16; m > 0; m >>= 1) s += __shfl_xor(s, m, 32);
    float l = mx + logf(s);
    out[(size_t)n * 64 + lane]      = v0 - l;
    out[(size_t)n * 64 + 32 + lane] = v1 - l;
}

// ---------------------------------------------------------------------------
// launch
// ---------------------------------------------------------------------------
extern "C" void kernel_launch(void* const* d_in, const int* in_sizes, int n_in,
                              void* d_out, int out_size, void* d_ws, size_t ws_size,
                              hipStream_t stream) {
    const float* x    = (const float*)d_in[0];
    const int*   ei   = (const int*)d_in[1];
    const float* W1   = (const float*)d_in[2];
    const float* b1   = (const float*)d_in[3];
    const float* W2   = (const float*)d_in[4];
    const float* b2   = (const float*)d_in[5];
    const float* wprj = (const float*)d_in[6];
    const float* bprj = (const float*)d_in[7];

    const int N = in_sizes[0] / 512;
    const int E = in_sizes[1] / 2;
    const int* row = ei;
    const int* col = ei + E;

    // scratch layout
    char* ws = (char*)d_ws;
    size_t off = 0;
    auto alloc = [&](size_t bytes) {
        size_t o = off;
        off = (off + bytes + 255) & ~(size_t)255;
        return o;
    };
    __bf16* w1hi = (__bf16*)(ws + alloc((size_t)512 * 256 * 2));
    __bf16* w1lo = (__bf16*)(ws + alloc((size_t)512 * 256 * 2));
    __bf16* w2hi = (__bf16*)(ws + alloc((size_t)256 * 64 * 2));
    __bf16* w2lo = (__bf16*)(ws + alloc((size_t)256 * 64 * 2));
    float*  dinv = (float*)(ws + alloc((size_t)N * 4));       // deg -> dinv in place
    float*  self = (float*)(ws + alloc((size_t)N * 4));
    float*  nrm  = (float*)(ws + alloc((size_t)E * 4));
    float*  hA   = (float*)(ws + alloc((size_t)N * 64 * 4));
    float*  hB   = (float*)(ws + alloc((size_t)N * 64 * 4));
    float*  oacc = (float*)(ws + alloc((size_t)N * 64 * 4));

    const long long NC = (long long)N * 64;

    // 1) weight split/transpose
    split_w_kernel<<<(512 * 256 + 255) / 256, 256, 0, stream>>>(W1, 512, 256, w1hi, w1lo);
    split_w_kernel<<<(256 * 64 + 255) / 256, 256, 0, stream>>>(W2, 256, 64, w2hi, w2lo);

    // 2) fused MLP -> hA
    mlp_kernel<<<(N + 15) / 16, 256, 0, stream>>>(x, w1hi, w1lo, b1, w2hi, w2lo, b2, hA, N);

    // 3) gcn_norm
    fill_kernel<<<(int)((N + 255) / 256), 256, 0, stream>>>(dinv, 1.0f, N);  // self-loop deg
    deg_kernel<<<(E + 255) / 256, 256, 0, stream>>>(col, dinv, E);
    dinv_kernel<<<(N + 255) / 256, 256, 0, stream>>>(dinv, self, N);
    norm_kernel<<<(E + 255) / 256, 256, 0, stream>>>(row, col, dinv, nrm, E);

    // 4) out accumulator, k=0 term
    fill_kernel<<<(int)((NC + 255) / 256), 256, 0, stream>>>(oacc, 0.0f, NC);
    retain_acc_kernel<<<(N + 7) / 8, 256, 0, stream>>>(hA, wprj, bprj, oacc, N);

    // 5) K=10 propagation rounds
    float* hp = hA;
    float* hn = hB;
    const long long EC = (long long)E * 64;
    for (int k = 0; k < 10; ++k) {
        selfinit_kernel<<<(int)((NC + 255) / 256), 256, 0, stream>>>(hp, self, hn, NC);
        scatter_kernel<<<(int)((EC + 255) / 256), 256, 0, stream>>>(row, col, nrm, hp, hn, E);
        retain_acc_kernel<<<(N + 7) / 8, 256, 0, stream>>>(hn, wprj, bprj, oacc, N);
        float* t = hp; hp = hn; hn = t;
    }

    // 6) log_softmax -> d_out
    lsm_kernel<<<(N + 7) / 8, 256, 0, stream>>>(oacc, (float*)d_out, N);
    (void)n_in; (void)out_size; (void)ws_size;
}
